// PILL_16655883174587
// MI455X (gfx1250) — matrix operand order repaired and checked by reading, not compiled, not run on hardware
//
#include <hip/hip_runtime.h>
#include <cstdint>
#include <cstddef>

#define NNODE   81920
#define NSESS   4096
#define PSEG    20
#define DDIM    128
#define DOUTDIM 256
#define MAXD    8

typedef __attribute__((ext_vector_type(16))) __bf16 v16bf;
typedef __attribute__((ext_vector_type(8)))  float  v8f;

union FragU { uint4 u[2]; v16bf v; };

// ---- CDNA5 WMMA fragment loaders (wave32 layouts from ISA 7.12.2) ----------

// A (16x32 bf16): lane l holds row M=l&15; dwords v=0..3 at row dwords
// base+0..3, v=4..7 at base+8..11, base = kc*16 + (l>=16 ? 4 : 0).
__device__ __forceinline__ v16bf load_afrag(const unsigned int* s, int kc, int lane) {
  int m = lane & 15, hi = lane >> 4;
  int base = m * 64 + kc * 16 + hi * 4;
  FragU f;
  f.u[0] = *(const uint4*)(s + base);
  f.u[1] = *(const uint4*)(s + base + 8);
  return f.v;
}

// B (32x16 bf16): pre-packed per (slot = jc*4+kc, lane): 16 contiguous halves.
__device__ __forceinline__ v16bf load_bfrag(const __bf16* buf, int slot, int lane) {
  const uint4* p = (const uint4*)(buf + (((size_t)slot * 32 + lane) << 4));
  FragU f; f.u[0] = p[0]; f.u[1] = p[1];
  return f.v;
}

// acc += X(16x128, split bf16 hi/lo staged in LDS) @ W^T col-tile jc,
// 3-product split-bf16 for ~fp32 accuracy: hi*hi + hi*lo + lo*hi.
__device__ __forceinline__ void mma_tile(v8f& acc,
                                         const unsigned int* xhi, const unsigned int* xlo,
                                         const __bf16* bhi, const __bf16* blo,
                                         int jc, int lane) {
#pragma unroll
  for (int kc = 0; kc < 4; ++kc) {
    v16bf ah = load_afrag(xhi, kc, lane);
    v16bf al = load_afrag(xlo, kc, lane);
    int slot = jc * 4 + kc;
    v16bf bh = load_bfrag(bhi, slot, lane);
    v16bf bl = load_bfrag(blo, slot, lane);
    acc = __builtin_amdgcn_wmma_f32_16x16x32_bf16(false, ah, false, bh, (short)0, acc, false, false);
    acc = __builtin_amdgcn_wmma_f32_16x16x32_bf16(false, ah, false, bl, (short)0, acc, false, false);
    acc = __builtin_amdgcn_wmma_f32_16x16x32_bf16(false, al, false, bh, (short)0, acc, false, false);
  }
}

__device__ __forceinline__ void split2(float x, __bf16* hi, __bf16* lo) {
  __bf16 h = (__bf16)x;
  *hi = h;
  *lo = (__bf16)(x - (float)h);
}

__device__ __forceinline__ float sigf(float x) { return 1.0f / (1.0f + expf(-x)); }

// ---- small utility kernels --------------------------------------------------

__global__ void k_zero(float* p, int n) {
  int i = blockIdx.x * blockDim.x + threadIdx.x;
  if (i < n) p[i] = 0.f;
}

__global__ __launch_bounds__(128) void k_colstats(const float* __restrict__ x,
                                                  float* __restrict__ sums) {
  int t = threadIdx.x;
  size_t r0 = (size_t)blockIdx.x * 128;
  float s = 0.f, q = 0.f;
  for (int r = 0; r < 128; ++r) {
    float v = x[(r0 + r) * DDIM + t];
    s += v; q += v * v;
  }
  atomicAdd(&sums[t], s);
  atomicAdd(&sums[128 + t], q);
}

__global__ void k_bnfin(const float* __restrict__ sums, const float* __restrict__ g,
                        const float* __restrict__ b, float* __restrict__ scale,
                        float* __restrict__ shift) {
  int t = threadIdx.x;
  if (t < 128) {
    float invn = 1.0f / (float)NNODE;
    float m = sums[t] * invn;
    float var = sums[128 + t] * invn - m * m;
    float is = rsqrtf(var + 1e-5f);
    float sc = g[t] * is;
    scale[t] = sc;
    shift[t] = b[t] - m * sc;
  }
}

__global__ __launch_bounds__(256) void k_bnapply(const float* __restrict__ x,
                                                 const float* __restrict__ scale,
                                                 const float* __restrict__ shift,
                                                 float* __restrict__ outF,
                                                 __bf16* __restrict__ oHi,
                                                 __bf16* __restrict__ oLo, int hasF) {
  size_t i = (size_t)blockIdx.x * 256 + threadIdx.x;
  if (i >= (size_t)NNODE * DDIM) return;
  int d = (int)(i & 127);
  float v = fmaf(x[i], scale[d], shift[d]);
  if (hasF) outF[i] = v;
  split2(v, &oHi[i], &oLo[i]);
}

// Repack W[C][128] (row-major, used as B = W^T) into per-lane B fragments.
__global__ __launch_bounds__(256) void k_repack(const float* __restrict__ w, int C,
                                                __bf16* __restrict__ dHi,
                                                __bf16* __restrict__ dLo) {
  int i = blockIdx.x * 256 + threadIdx.x;
  if (i >= C * DDIM) return;
  int c = i >> 7, k = i & 127;
  int jc = c >> 4, ncol = c & 15, kc = k >> 5, kk = k & 31;
  int hi = (kk >> 3) & 1, e = (kk & 7) + ((kk >> 4) << 3);
  size_t dst = (((size_t)(jc * 4 + kc) * 32 + (ncol + (hi << 4))) << 4) + e;
  split2(w[i], &dHi[dst], &dLo[dst]);
}

// ---- GRU over neighbor mailboxes + rst epilogue (1 wave = 16 nodes) --------
// All recurrent state (h fp32, z, r) is LDS-resident to keep VGPR pressure low
// (round-1 build spilled hoisted weight fragments to scratch at ~1024 VGPRs).

__global__ __launch_bounds__(32) __attribute__((amdgpu_waves_per_eu(2)))
void k_gru(
    const int* __restrict__ nbr, const int* __restrict__ deg,
    const __bf16* __restrict__ h0hi, const __bf16* __restrict__ h0lo,
    const __bf16* __restrict__ wihHi, const __bf16* __restrict__ wihLo,
    const __bf16* __restrict__ whhHi, const __bf16* __restrict__ whhLo,
    const __bf16* __restrict__ wsHi,  const __bf16* __restrict__ wsLo,
    const __bf16* __restrict__ wnHi,  const __bf16* __restrict__ wnLo,
    const float* __restrict__ bih, const float* __restrict__ bhh,
    float* __restrict__ rst) {
  __shared__ __align__(16) unsigned int sXhi[1024], sXlo[1024], sHhi[1024], sHlo[1024];
  __shared__ float sR[2048], sZ[2048], sHf[2048];
  int lane = threadIdx.x, tile = blockIdx.x;
  int n = lane & 15, hi8 = (lane >> 4) << 3;
  const uint4* gHi = (const uint4*)h0hi;
  const uint4* gLo = (const uint4*)h0lo;

  int degv[8];
#pragma unroll
  for (int v = 0; v < 8; ++v) degv[v] = deg[tile * 16 + v + hi8];

  for (int i = lane; i < 1024; i += 32) { sHhi[i] = 0u; sHlo[i] = 0u; }
  for (int i = lane; i < 2048; i += 32) sHf[i] = 0.f;
  __syncthreads();

  for (int t = 0; t < MAXD; ++t) {
    asm volatile("" ::: "memory");   // block cross-step hoisting of weight frags
    // gather x_t = h0[nbr_idx[:, t]] rows (hi/lo) into LDS, 16B chunks
    for (int i = lane; i < 256; i += 32) {
      int r = i >> 4, c = i & 15;
      size_t idx = (size_t)nbr[(tile * 16 + r) * MAXD + t];
      ((uint4*)sXhi)[i] = gHi[idx * 16 + c];
      ((uint4*)sXlo)[i] = gLo[idx * 16 + c];
    }
    __syncthreads();

#pragma clang loop unroll(disable)
    for (int j = 0; j < 8; ++j) {               // z = sigmoid(iz + hz) -> LDS
      float bb = bih[DDIM + j * 16 + n] + bhh[DDIM + j * 16 + n];
      v8f acc;
#pragma unroll
      for (int v = 0; v < 8; ++v) acc[v] = bb;
      mma_tile(acc, sXhi, sXlo, wihHi, wihLo, 8 + j, lane);
      mma_tile(acc, sHhi, sHlo, whhHi, whhLo, 8 + j, lane);
#pragma unroll
      for (int v = 0; v < 8; ++v) sZ[(v + hi8) * 128 + j * 16 + n] = sigf(acc[v]);
    }
#pragma clang loop unroll(disable)
    for (int j = 0; j < 8; ++j) {               // r = sigmoid(ir + hr) -> LDS
      float bb = bih[j * 16 + n] + bhh[j * 16 + n];
      v8f acc;
#pragma unroll
      for (int v = 0; v < 8; ++v) acc[v] = bb;
      mma_tile(acc, sXhi, sXlo, wihHi, wihLo, j, lane);
      mma_tile(acc, sHhi, sHlo, whhHi, whhLo, j, lane);
#pragma unroll
      for (int v = 0; v < 8; ++v) sR[(v + hi8) * 128 + j * 16 + n] = sigf(acc[v]);
    }
    __syncthreads();
#pragma clang loop unroll(disable)
    for (int j = 0; j < 8; ++j) {               // n gate + update + freeze mask
      float bbi = bih[2 * DDIM + j * 16 + n], bbh = bhh[2 * DDIM + j * 16 + n];
      v8f ai, ah;
#pragma unroll
      for (int v = 0; v < 8; ++v) { ai[v] = bbi; ah[v] = bbh; }
      mma_tile(ai, sXhi, sXlo, wihHi, wihLo, 16 + j, lane);
      mma_tile(ah, sHhi, sHlo, whhHi, whhLo, 16 + j, lane);
#pragma unroll
      for (int v = 0; v < 8; ++v) {
        int addr = (v + hi8) * 128 + j * 16 + n;
        float r  = sR[addr];
        float z  = sZ[addr];
        float hp = sHf[addr];
        float nn = tanhf(ai[v] + r * ah[v]);
        float hn = (1.f - z) * nn + z * hp;
        sHf[addr] = (t < degv[v]) ? hn : hp;    // old h frags in sHhi/sHlo intact
      }
    }
    __syncthreads();
    // restage h (hi/lo) for next step's A-fragments
    for (int i = lane; i < 2048; i += 32)
      split2(sHf[i], &((__bf16*)sHhi)[i], &((__bf16*)sHlo)[i]);
    __syncthreads();
  }

  // rst = h0 @ W_self^T + neigh @ W_neigh^T (neigh still staged in sH)
  for (int i = lane; i < 256; i += 32) {
    ((uint4*)sXhi)[i] = gHi[(size_t)tile * 256 + i];
    ((uint4*)sXlo)[i] = gLo[(size_t)tile * 256 + i];
  }
  __syncthreads();
#pragma clang loop unroll(disable)
  for (int j = 0; j < 8; ++j) {
    v8f acc;
#pragma unroll
    for (int v = 0; v < 8; ++v) acc[v] = 0.f;
    mma_tile(acc, sXhi, sXlo, wsHi, wsLo, j, lane);
    mma_tile(acc, sHhi, sHlo, wnHi, wnLo, j, lane);
#pragma unroll
    for (int v = 0; v < 8; ++v)
      rst[((size_t)(tile * 16 + v + hi8)) * DDIM + j * 16 + n] = acc[v];
  }
}

// ---- fu = h @ Wu^T (WMMA) ---------------------------------------------------

__global__ __launch_bounds__(32) void k_fu(const __bf16* __restrict__ hhi,
                                           const __bf16* __restrict__ hlo,
                                           const __bf16* __restrict__ wuHi,
                                           const __bf16* __restrict__ wuLo,
                                           float* __restrict__ fu) {
  __shared__ __align__(16) unsigned int sXhi[1024], sXlo[1024];
  int lane = threadIdx.x, tile = blockIdx.x;
  int n = lane & 15, hi8 = (lane >> 4) << 3;
  const uint4* gHi = (const uint4*)hhi;
  const uint4* gLo = (const uint4*)hlo;
  for (int i = lane; i < 256; i += 32) {
    ((uint4*)sXhi)[i] = gHi[(size_t)tile * 256 + i];
    ((uint4*)sXlo)[i] = gLo[(size_t)tile * 256 + i];
  }
  __syncthreads();
#pragma clang loop unroll(disable)
  for (int j = 0; j < 8; ++j) {
    v8f acc;
#pragma unroll
    for (int v = 0; v < 8; ++v) acc[v] = 0.f;
    mma_tile(acc, sXhi, sXlo, wuHi, wuLo, j, lane);
#pragma unroll
    for (int v = 0; v < 8; ++v)
      fu[((size_t)(tile * 16 + v + hi8)) * DDIM + j * 16 + n] = acc[v];
  }
}

// ---- per-session attention: fv/fl, e, softmax, segment sums ----------------

__global__ __launch_bounds__(128) void k_attn(
    const float* __restrict__ fu, const float* __restrict__ hfull,
    const float* __restrict__ intend, const int* __restrict__ last,
    const float* __restrict__ posw,
    const float* __restrict__ Wv, const float* __restrict__ bv,
    const float* __restrict__ Wi, const float* __restrict__ bi,
    const float* __restrict__ We,
    __bf16* __restrict__ s1hi, __bf16* __restrict__ s1lo,
    __bf16* __restrict__ s2hi, __bf16* __restrict__ s2lo) {
  __shared__ float xin[128], xl[128], comb[128], wev[128], ebuf[PSEG], alpha[PSEG], pw[PSEG];
  int b = blockIdx.x, t = threadIdx.x;
  xin[t] = intend[(size_t)b * DDIM + t];
  wev[t] = We[t];
  int ln = last[b];
  xl[t] = hfull[(size_t)ln * DDIM + t];
  if (t < PSEG) pw[t] = posw[(size_t)b * PSEG + t];
  __syncthreads();
  {
    float a1 = bv[t], a2 = bi[t];
    const float* wv = Wv + (size_t)t * DDIM;
    const float* wi = Wi + (size_t)t * DDIM;
    for (int k = 0; k < DDIM; ++k) {
      a1 = fmaf(xin[k], wv[k], a1);
      a2 = fmaf(xl[k],  wi[k], a2);
    }
    comb[t] = a1 + a2;
  }
  __syncthreads();
  int w = t >> 5, lane = t & 31;       // 4 waves x 5 positions = 20
#pragma unroll
  for (int pp = 0; pp < 5; ++pp) {
    int p = w * 5 + pp;
    const float* fup = fu + ((size_t)b * PSEG + p) * DDIM;
    float part = 0.f;
    for (int d = lane; d < DDIM; d += 32) part += wev[d] * sigf(fup[d] + comb[d]);
    for (int off = 16; off > 0; off >>= 1) part += __shfl_xor(part, off, 32);
    if (lane == 0) ebuf[p] = part;
  }
  __syncthreads();
  if (t == 0) {
    float mx = ebuf[0];
    for (int p = 1; p < PSEG; ++p) mx = fmaxf(mx, ebuf[p]);
    float s = 0.f;
    for (int p = 0; p < PSEG; ++p) { float ex = expf(ebuf[p] - mx); alpha[p] = ex; s += ex; }
    float inv = 1.f / s;
    for (int p = 0; p < PSEG; ++p) alpha[p] *= inv;
  }
  __syncthreads();
  {
    float s1 = 0.f, s2 = 0.f;
    for (int p = 0; p < PSEG; ++p) {
      float hv = hfull[((size_t)b * PSEG + p) * DDIM + t];
      s1 = fmaf(alpha[p], hv, s1);
      s2 = fmaf(pw[p],    hv, s2);
    }
    split2(s1, &s1hi[(size_t)b * DDIM + t], &s1lo[(size_t)b * DDIM + t]);
    split2(s2, &s2hi[(size_t)b * DDIM + t], &s2lo[(size_t)b * DDIM + t]);
  }
}

// ---- [s1; s2] @ Wout^T -> d_out (WMMA) -------------------------------------

__global__ __launch_bounds__(32) void k_out(
    const __bf16* __restrict__ s1hi, const __bf16* __restrict__ s1lo,
    const __bf16* __restrict__ s2hi, const __bf16* __restrict__ s2lo,
    const __bf16* __restrict__ woHi, const __bf16* __restrict__ woLo,
    float* __restrict__ out) {
  __shared__ __align__(16) unsigned int sXhi[1024], sXlo[1024];
  int lane = threadIdx.x;
  int which = blockIdx.x >> 8;   // 0 = rst_g, 1 = pos
  int rt = blockIdx.x & 255;
  int n = lane & 15, hi8 = (lane >> 4) << 3;
  const uint4* gHi = (const uint4*)(which ? s2hi : s1hi);
  const uint4* gLo = (const uint4*)(which ? s2lo : s1lo);
  for (int i = lane; i < 256; i += 32) {
    ((uint4*)sXhi)[i] = gHi[(size_t)rt * 256 + i];
    ((uint4*)sXlo)[i] = gLo[(size_t)rt * 256 + i];
  }
  __syncthreads();
  float* o = out + (size_t)which * NSESS * DOUTDIM;
#pragma clang loop unroll(disable)
  for (int j = 0; j < 16; ++j) {
    v8f acc;
#pragma unroll
    for (int v = 0; v < 8; ++v) acc[v] = 0.f;
    mma_tile(acc, sXhi, sXlo, woHi, woLo, j, lane);
#pragma unroll
    for (int v = 0; v < 8; ++v)
      o[((size_t)(rt * 16 + v + hi8)) * DOUTDIM + j * 16 + n] = acc[v];
  }
}

// ---- host orchestration -----------------------------------------------------

extern "C" void kernel_launch(void* const* d_in, const int* in_sizes, int n_in,
                              void* d_out, int out_size, void* d_ws, size_t ws_size,
                              hipStream_t stream) {
  (void)in_sizes; (void)n_in; (void)out_size; (void)ws_size;
  const float* feat   = (const float*)d_in[0];
  const float* intend = (const float*)d_in[1];
  const float* posw   = (const float*)d_in[2];
  const int*   nbr    = (const int*)d_in[3];
  const int*   deg    = (const int*)d_in[4];
  const int*   last   = (const int*)d_in[5];
  const float* bn1g = (const float*)d_in[6];
  const float* bn1b = (const float*)d_in[7];
  const float* Wih  = (const float*)d_in[8];
  const float* Whh  = (const float*)d_in[9];
  const float* bih  = (const float*)d_in[10];
  const float* bhh  = (const float*)d_in[11];
  const float* Wself  = (const float*)d_in[12];
  const float* Wneigh = (const float*)d_in[13];
  const float* bn2g = (const float*)d_in[14];
  const float* bn2b = (const float*)d_in[15];
  const float* Wu   = (const float*)d_in[16];
  const float* Wv   = (const float*)d_in[17];
  const float* bvv  = (const float*)d_in[18];
  const float* Wi   = (const float*)d_in[19];
  const float* bii  = (const float*)d_in[20];
  const float* We   = (const float*)d_in[21];
  const float* Wout = (const float*)d_in[22];

  char* W = (char*)d_ws;
  const size_t ND = (size_t)NNODE * DDIM;   // 10,485,760 elements
  const size_t BD = (size_t)NSESS * DDIM;   // 524,288 elements

  // stats block (floats): [sum1,sq1 | sc1,sh1 | sum2,sq2 | sc2,sh2]
  float* st = (float*)W;
  // repacked weights
  size_t o = 4096;
  __bf16* wihHi = (__bf16*)(W + o); o += 384 * 128 * 2;
  __bf16* wihLo = (__bf16*)(W + o); o += 384 * 128 * 2;
  __bf16* whhHi = (__bf16*)(W + o); o += 384 * 128 * 2;
  __bf16* whhLo = (__bf16*)(W + o); o += 384 * 128 * 2;
  __bf16* wsHi  = (__bf16*)(W + o); o += 128 * 128 * 2;
  __bf16* wsLo  = (__bf16*)(W + o); o += 128 * 128 * 2;
  __bf16* wnHi  = (__bf16*)(W + o); o += 128 * 128 * 2;
  __bf16* wnLo  = (__bf16*)(W + o); o += 128 * 128 * 2;
  __bf16* wuHi  = (__bf16*)(W + o); o += 128 * 128 * 2;
  __bf16* wuLo  = (__bf16*)(W + o); o += 128 * 128 * 2;
  __bf16* woHi  = (__bf16*)(W + o); o += 256 * 128 * 2;
  __bf16* woLo  = (__bf16*)(W + o); o += 256 * 128 * 2;
  // big buffers
  size_t A = 1u << 20;
  __bf16* aHi = (__bf16*)(W + A);                    // h0 hi, later h hi
  __bf16* aLo = (__bf16*)(W + A + ND * 2);           // h0 lo, later h lo
  float*  rstF = (float*)(W + A + ND * 4);           // rst, later fu
  float*  hF   = (float*)(W + A + ND * 8);           // h (post bn2) fp32
  size_t S = A + ND * 12;
  __bf16* s1hi = (__bf16*)(W + S);
  __bf16* s1lo = (__bf16*)(W + S + BD * 2);
  __bf16* s2hi = (__bf16*)(W + S + BD * 4);
  __bf16* s2lo = (__bf16*)(W + S + BD * 6);

  // 0) zero stat accumulators
  k_zero<<<4, 256, 0, stream>>>(st, 1024);
  // 1) bn1 stats + apply (-> h0 hi/lo)
  k_colstats<<<NNODE / 128, 128, 0, stream>>>(feat, st);
  k_bnfin<<<1, 128, 0, stream>>>(st, bn1g, bn1b, st + 256, st + 384);
  k_bnapply<<<(unsigned)(ND / 256), 256, 0, stream>>>(feat, st + 256, st + 384,
                                                      (float*)nullptr, aHi, aLo, 0);
  // 2) repack weights into B-fragment layout
  k_repack<<<192, 256, 0, stream>>>(Wih, 384, wihHi, wihLo);
  k_repack<<<192, 256, 0, stream>>>(Whh, 384, whhHi, whhLo);
  k_repack<<<64, 256, 0, stream>>>(Wself, 128, wsHi, wsLo);
  k_repack<<<64, 256, 0, stream>>>(Wneigh, 128, wnHi, wnLo);
  k_repack<<<64, 256, 0, stream>>>(Wu, 128, wuHi, wuLo);
  k_repack<<<128, 256, 0, stream>>>(Wout, 256, woHi, woLo);
  // 3) GRU + rst epilogue (dominant WMMA work)
  k_gru<<<NNODE / 16, 32, 0, stream>>>(nbr, deg, aHi, aLo,
                                       wihHi, wihLo, whhHi, whhLo,
                                       wsHi, wsLo, wnHi, wnLo, bih, bhh, rstF);
  // 4) bn2 stats + apply (-> h fp32 + h hi/lo, reusing h0 buffers)
  k_colstats<<<NNODE / 128, 128, 0, stream>>>(rstF, st + 512);
  k_bnfin<<<1, 128, 0, stream>>>(st + 512, bn2g, bn2b, st + 768, st + 896);
  k_bnapply<<<(unsigned)(ND / 256), 256, 0, stream>>>(rstF, st + 768, st + 896,
                                                      hF, aHi, aLo, 1);
  // 5) fu = h @ Wu^T (overwrites rst buffer)
  k_fu<<<NNODE / 16, 32, 0, stream>>>(aHi, aLo, wuHi, wuLo, rstF);
  // 6) attention: e, softmax, segment sums -> s1/s2 (bf16 hi/lo)
  k_attn<<<NSESS, 128, 0, stream>>>(rstF, hF, intend, last, posw,
                                    Wv, bvv, Wi, bii, We, s1hi, s1lo, s2hi, s2lo);
  // 7) [s1;s2] @ Wout^T -> d_out (rst_g then pos)
  k_out<<<2 * (NSESS / 16), 32, 0, stream>>>(s1hi, s1lo, s2hi, s2lo,
                                             woHi, woLo, (float*)d_out);
}
